// D_GCN_47227460387502
// MI455X (gfx1250) — matrix operand order repaired and checked by reading, not compiled, not run on hardware
//
#include <hip/hip_runtime.h>
#include <hip/hip_bf16.h>
#include <float.h>

// ---------------------------------------------------------------------------
// Problem constants (match reference)
// ---------------------------------------------------------------------------
#define NN 8192
#define FF 128
#define HH 128
#define KK 16
#define EE (NN * 32)          // 262144 edges
#define ET (EE + NN)          // edges + self loops

typedef __attribute__((ext_vector_type(16))) _Float16 v16h;
typedef __attribute__((ext_vector_type(8)))  _Float16 v8h;
typedef __attribute__((ext_vector_type(8)))  float    v8f;

// ---------------------------------------------------------------------------
// WMMA helpers (CDNA5: v_wmma_f32_16x16x32_f16, wave32)
// ---------------------------------------------------------------------------
__device__ __forceinline__ v8f wmma16(v16h a, v16h b, v8f c) {
  // (neg_a, A, neg_b, B, c_mod, C, reuse_a, reuse_b)
  return __builtin_amdgcn_wmma_f32_16x16x32_f16(false, a, false, b, (short)0, c,
                                                false, false);
}

// A tile 16xK(32) from row-major storage, lda = row stride (elements).
// Lanes 0-15 row M=lane: halves 0-15 -> K=k0+hi*16+0..15
__device__ __forceinline__ v16h load_a(const _Float16* A, int lda, int k0) {
  int lane = threadIdx.x & 31;
  int r = lane & 15, hi = lane >> 4;
  const _Float16* p = A + (size_t)r * lda + k0 + hi * 16;
  v16h a;
#pragma unroll
  for (int t = 0; t < 8; t++) a[t] = p[t];
#pragma unroll
  for (int t = 0; t < 8; t++) a[8 + t] = p[8 + t];
  return a;
}

// B tile K(32)x16 where B is stored TRANSPOSED row-major: Bt[n][k], stride ldx.
// Lane holds column n0+(lane&15), K range k0+hi*16 .. +15 -> 16 contiguous
// halves per lane (32B) => two global_load_b128.
__device__ __forceinline__ v16h load_bt(const _Float16* Xt, int ldx, int k0, int n0) {
  int lane = threadIdx.x & 31;
  int c = lane & 15, hi = lane >> 4;
  const _Float16* p = Xt + (size_t)(n0 + c) * ldx + k0 + hi * 16;
  v16h b;
#pragma unroll
  for (int t = 0; t < 16; t++) b[t] = p[t];
  return b;
}

// ---------------------------------------------------------------------------
// Small utility kernels
// ---------------------------------------------------------------------------
__global__ void k_zero(float* p, int n) {
  int i = blockIdx.x * blockDim.x + threadIdx.x;
  if (i < n) p[i] = 0.0f;
}

__global__ void k_f32_to_h(const float* in, _Float16* out, int n) {
  int i = blockIdx.x * blockDim.x + threadIdx.x;
  if (i < n) out[i] = (_Float16)in[i];
}

// f32 [K][N] -> f16 transposed [N][K]
__global__ void k_f32_to_h_T(const float* in, _Float16* out, int K, int N) {
  int i = blockIdx.x * blockDim.x + threadIdx.x;
  if (i < K * N) {
    int k = i / N, n = i % N;
    out[(size_t)n * K + k] = (_Float16)in[i];
  }
}

// degree accumulation over edges + self loops
__global__ void k_deg(const int* dst, const float* ew, float* deg) {
  int e = blockIdx.x * blockDim.x + threadIdx.x;
  if (e < EE)      atomicAdd(&deg[dst[e]], ew[e]);
  else if (e < ET) atomicAdd(&deg[e - EE], 1.0f);
}

__global__ void k_dis(const float* deg, float* dis) {
  int i = blockIdx.x * blockDim.x + threadIdx.x;
  if (i < NN) {
    float d = deg[i];
    dis[i] = (d > 0.0f) ? rsqrtf(fmaxf(d, 1e-12f)) : 0.0f;
  }
}

__global__ void k_norm(const int* src, const int* dst, const float* ew,
                       const float* dis, float* norm) {
  int e = blockIdx.x * blockDim.x + threadIdx.x;
  if (e < EE)      norm[e] = dis[src[e]] * ew[e] * dis[dst[e]];
  else if (e < ET) { float d = dis[e - EE]; norm[e] = d * d; }
}

// out[dst] += norm * h[src], 128 feats/edge, 4 feats per lane (32 lanes/edge)
__global__ void k_scatter(const float* __restrict__ h, const float* __restrict__ norm,
                          const int* __restrict__ src, const int* __restrict__ dst,
                          float* out) {
  long long gid = (long long)blockIdx.x * blockDim.x + threadIdx.x;
  int e = (int)(gid >> 5);
  if (e >= ET) return;
  int c0 = ((int)gid & 31) * 4;
  int s, d;
  if (e < EE) { s = src[e]; d = dst[e]; } else { s = d = e - EE; }
  float w = norm[e];
  const float* hp = h + (size_t)s * HH + c0;
  float* op = out + (size_t)d * HH + c0;
#pragma unroll
  for (int q = 0; q < 4; q++) atomicAdd(&op[q], w * hp[q]);
}

__global__ void k_bias_tanh(const float* agg, const float* b, float* nf, _Float16* nh) {
  int i = blockIdx.x * blockDim.x + threadIdx.x;
  if (i < NN * HH) {
    float v = tanhf(agg[i] + b[i & (HH - 1)]);
    nf[i] = v;
    nh[i] = (_Float16)v;
  }
}

__global__ void k_sn(const float* X, float* sn) {
  int row = blockIdx.x * (blockDim.x >> 5) + (threadIdx.x >> 5);
  int lane = threadIdx.x & 31;
  float s = 0.0f;
#pragma unroll
  for (int q = 0; q < 4; q++) {
    float v = X[(size_t)row * HH + lane + q * 32];
    s += v * v;
  }
#pragma unroll
  for (int o = 16; o; o >>= 1) s += __shfl_xor(s, o, 32);
  if (lane == 0) sn[row] = s;
}

// s1[row] = || e + n ||_2
__global__ void k_rownorm(const float* e, const float* n, float* s1) {
  int row = blockIdx.x * (blockDim.x >> 5) + (threadIdx.x >> 5);
  int lane = threadIdx.x & 31;
  float s = 0.0f;
#pragma unroll
  for (int q = 0; q < 4; q++) {
    size_t i = (size_t)row * HH + lane + q * 32;
    float v = e[i] + n[i];
    s += v * v;
  }
#pragma unroll
  for (int o = 16; o; o >>= 1) s += __shfl_xor(s, o, 32);
  if (lane == 0) s1[row] = sqrtf(s);
}

__global__ void k_add(const float* e, const float* n, float* t) {
  int i = blockIdx.x * blockDim.x + threadIdx.x;
  if (i < NN * HH) t[i] = e[i] + n[i];
}

// x2 = s1 * t2 * (e3 + n3); log_softmax per row -> out
__global__ void k_final(const float* e3, const float* n3, const float* t2,
                        const float* s1, float* out) {
  int row = blockIdx.x * (blockDim.x >> 5) + (threadIdx.x >> 5);
  int lane = threadIdx.x & 31;
  float sc = s1[row];
  float v[4];
  float m = -FLT_MAX;
#pragma unroll
  for (int q = 0; q < 4; q++) {
    size_t i = (size_t)row * HH + lane + q * 32;
    v[q] = sc * t2[i] * (e3[i] + n3[i]);
    m = fmaxf(m, v[q]);
  }
#pragma unroll
  for (int o = 16; o; o >>= 1) m = fmaxf(m, __shfl_xor(m, o, 32));
  float s = 0.0f;
#pragma unroll
  for (int q = 0; q < 4; q++) s += __expf(v[q] - m);
#pragma unroll
  for (int o = 16; o; o >>= 1) s += __shfl_xor(s, o, 32);
  float lse = __logf(s);
#pragma unroll
  for (int q = 0; q < 4; q++)
    out[(size_t)row * HH + lane + q * 32] = v[q] - m - lse;
}

// ---------------------------------------------------------------------------
// WMMA GEMM row-strip: C[8192,128] = A[8192,128] @ W[128,128].
// Wt is W transposed f16 [n][k]. One wave computes a 16x128 strip:
// A fragments hoisted once, reused across 8 N-tiles.
// ---------------------------------------------------------------------------
__global__ void k_gemm_strip(const _Float16* __restrict__ A,
                             const _Float16* __restrict__ Wt,
                             float* __restrict__ C) {
  int wave = blockIdx.x * (blockDim.x >> 5) + (threadIdx.x >> 5);
  int row0 = wave * 16;
  const _Float16* At = A + (size_t)row0 * HH;
  v16h a0 = load_a(At, HH, 0);
  v16h a1 = load_a(At, HH, 32);
  v16h a2 = load_a(At, HH, 64);
  v16h a3 = load_a(At, HH, 96);
  int lane = threadIdx.x & 31, hi = lane >> 4, col = lane & 15;
#pragma unroll
  for (int nt = 0; nt < 8; nt++) {
    v8f c = {};
    c = wmma16(a0, load_bt(Wt, HH, 0,  nt * 16), c);
    c = wmma16(a1, load_bt(Wt, HH, 32, nt * 16), c);
    c = wmma16(a2, load_bt(Wt, HH, 64, nt * 16), c);
    c = wmma16(a3, load_bt(Wt, HH, 96, nt * 16), c);
#pragma unroll
    for (int r = 0; r < 8; r++)
      C[(size_t)(row0 + r + 8 * hi) * HH + nt * 16 + col] = c[r];
  }
}

// ---------------------------------------------------------------------------
// Fused kNN: stream -2*X@X^T via WMMA, 64 columns per barrier pair,
// per-row sorted top-16 kept in registers (lanes 0..15 own one row each).
// blockDim = 128 (4 waves); each wave owns 16 rows.
// ---------------------------------------------------------------------------
__global__ void k_knn(const _Float16* __restrict__ Xh, const float* __restrict__ sn,
                      int* __restrict__ knn_out) {
  __shared__ float dtile[4][16][68];      // stride 68 -> conflict-free row scan
  int wl = threadIdx.x >> 5;
  int wave = blockIdx.x * 4 + wl;
  int row0 = wave * 16;
  int lane = threadIdx.x & 31;
  int hi = lane >> 4, col = lane & 15;

  const _Float16* Arow = Xh + (size_t)row0 * HH;
  v16h a0 = load_a(Arow, HH, 0);
  v16h a1 = load_a(Arow, HH, 32);
  v16h a2 = load_a(Arow, HH, 64);
  v16h a3 = load_a(Arow, HH, 96);

  float sni = (lane < 16) ? sn[row0 + lane] : 0.0f;

  float bestv[KK];
  int   besti[KK];
#pragma unroll
  for (int s = 0; s < KK; s++) { bestv[s] = FLT_MAX; besti[s] = 0; }

  for (int cb = 0; cb < NN; cb += 64) {
    v8f acc[4];
#pragma unroll
    for (int t = 0; t < 4; t++) {
      int nb = cb + t * 16;
      v8f c = {};
      c = wmma16(a0, load_bt(Xh, HH, 0,  nb), c);
      c = wmma16(a1, load_bt(Xh, HH, 32, nb), c);
      c = wmma16(a2, load_bt(Xh, HH, 64, nb), c);
      c = wmma16(a3, load_bt(Xh, HH, 96, nb), c);
      acc[t] = c;
    }

    __syncthreads();                      // previous scan done before overwrite
#pragma unroll
    for (int t = 0; t < 4; t++) {
      float snj = sn[cb + t * 16 + col];
#pragma unroll
      for (int r = 0; r < 8; r++)
        dtile[wl][r + 8 * hi][t * 16 + col] = snj - 2.0f * acc[t][r];
    }
    __syncthreads();                      // tiles visible to scanning lanes

    if (lane < 16) {
      for (int j = 0; j < 64; j++) {
        float d = sni + dtile[wl][lane][j];
        int id = cb + j;
        if (d < bestv[KK - 1]) {          // beats current worst -> bubble-insert
#pragma unroll
          for (int s = 0; s < KK; s++) {
            if (d < bestv[s]) {
              float tv = bestv[s]; bestv[s] = d; d = tv;
              int   ti = besti[s]; besti[s] = id; id = ti;
            }
          }
        }
      }
    }
  }

  if (lane < 16) {
#pragma unroll
    for (int s = 0; s < KK; s++)
      knn_out[(size_t)(row0 + lane) * KK + s] = besti[s];
  }
}

// ---------------------------------------------------------------------------
// EdgeConv: one wave per node; 16 neighbors = one WMMA M-tile, so the
// max-over-k aggregation is a within-tile column max.
// f = [x_i, x_j - x_i] (16x256) -> relu@w1 (16x64) -> relu@w2 (16x128)
// -> @w3 (16x128) -> colmax + b3.  Weights pre-transposed f16 [N][K].
// blockDim = 128 (4 waves).
// ---------------------------------------------------------------------------
__global__ void k_edgeconv(const _Float16* __restrict__ Xh, const int* __restrict__ knn,
                           const _Float16* __restrict__ w1t, const float* __restrict__ b1,
                           const _Float16* __restrict__ w2t, const float* __restrict__ b2,
                           const _Float16* __restrict__ w3t, const float* __restrict__ b3,
                           float* __restrict__ Eout) {
  __shared__ __align__(16) _Float16 fbuf[4][16 * 256]; // f, reused for h2 (16x128)
  __shared__ __align__(16) _Float16 hbuf[4][16 * 64];  // h1
  int wl = threadIdx.x >> 5;
  int node = blockIdx.x * 4 + wl;
  int lane = threadIdx.x & 31;
  int hi = lane >> 4, col = lane & 15;

  // build f (16x256 f16) in LDS: 512 8-half chunks, vectorized 16B loads/stores
  const v8h* Xi = (const v8h*)(Xh + (size_t)node * HH);  // 16 chunks of 8 halves
  for (int t = lane; t < 512; t += 32) {
    int j = t >> 5, cc = t & 31;
    v8h v;
    if (cc < 16) {
      v = Xi[cc];
    } else {
      int nb = knn[(size_t)node * KK + j];
      const v8h* Xj = (const v8h*)(Xh + (size_t)nb * HH);
      v = Xj[cc - 16] - Xi[cc - 16];
    }
    *(v8h*)&fbuf[wl][j * 256 + cc * 8] = v;
  }
  __syncthreads();

  // layer 1: (16x256)@(256x64) -> relu -> h1 f16
  const _Float16* fA = &fbuf[wl][0];
#pragma unroll
  for (int nt = 0; nt < 4; nt++) {
    v8f c = {};
#pragma unroll
    for (int k = 0; k < 256; k += 32)
      c = wmma16(load_a(fA, 256, k), load_bt(w1t, 256, k, nt * 16), c);
    float bb = b1[nt * 16 + col];
#pragma unroll
    for (int r = 0; r < 8; r++) {
      float v = fmaxf(c[r] + bb, 0.0f);
      hbuf[wl][(r + 8 * hi) * 64 + nt * 16 + col] = (_Float16)v;
    }
  }
  __syncthreads();

  // layer 2: (16x64)@(64x128) -> relu -> h2 f16 (reuse fbuf as 16x128)
  const _Float16* hA = &hbuf[wl][0];
#pragma unroll
  for (int nt = 0; nt < 8; nt++) {
    v8f c = {};
    c = wmma16(load_a(hA, 64, 0),  load_bt(w2t, 64, 0,  nt * 16), c);
    c = wmma16(load_a(hA, 64, 32), load_bt(w2t, 64, 32, nt * 16), c);
    float bb = b2[nt * 16 + col];
#pragma unroll
    for (int r = 0; r < 8; r++) {
      float v = fmaxf(c[r] + bb, 0.0f);
      fbuf[wl][(r + 8 * hi) * 128 + nt * 16 + col] = (_Float16)v;
    }
  }
  __syncthreads();

  // layer 3: (16x128)@(128x128); column max over the 16 neighbor rows + b3
  const _Float16* gA = &fbuf[wl][0];
#pragma unroll
  for (int nt = 0; nt < 8; nt++) {
    v8f c = {};
#pragma unroll
    for (int k = 0; k < 128; k += 32)
      c = wmma16(load_a(gA, 128, k), load_bt(w3t, 128, k, nt * 16), c);
    float m = c[0];
#pragma unroll
    for (int r = 1; r < 8; r++) m = fmaxf(m, c[r]);
    m = fmaxf(m, __shfl_xor(m, 16, 32));  // combine row halves (same column)
    if (lane < 16)
      Eout[(size_t)node * HH + nt * 16 + lane] = m + b3[nt * 16 + lane];
  }
}

// ---------------------------------------------------------------------------
// Host-side orchestration
// ---------------------------------------------------------------------------
static inline size_t alignup(size_t x) { return (x + 255) & ~(size_t)255; }

extern "C" void kernel_launch(void* const* d_in, const int* in_sizes, int n_in,
                              void* d_out, int out_size, void* d_ws, size_t ws_size,
                              hipStream_t stream) {
  (void)in_sizes; (void)n_in; (void)out_size; (void)ws_size;

  const float* x  = (const float*)d_in[0];
  const int*   ei = (const int*)d_in[1];
  const float* ew = (const float*)d_in[2];
  const int* src = ei;
  const int* dst = ei + EE;

  const float* wg[3]  = { (const float*)d_in[3], (const float*)d_in[5], (const float*)d_in[7] };
  const float* bg[3]  = { (const float*)d_in[4], (const float*)d_in[6], (const float*)d_in[8] };
  const float* ew1[3] = { (const float*)d_in[9],  (const float*)d_in[15], (const float*)d_in[21] };
  const float* eb1[3] = { (const float*)d_in[10], (const float*)d_in[16], (const float*)d_in[22] };
  const float* ew2[3] = { (const float*)d_in[11], (const float*)d_in[17], (const float*)d_in[23] };
  const float* eb2[3] = { (const float*)d_in[12], (const float*)d_in[18], (const float*)d_in[24] };
  const float* ew3[3] = { (const float*)d_in[13], (const float*)d_in[19], (const float*)d_in[25] };
  const float* eb3[3] = { (const float*)d_in[14], (const float*)d_in[20], (const float*)d_in[26] };

  // ---- workspace carve-up ----
  char* base = (char*)d_ws;
  size_t off = 0;
  auto carve = [&](size_t bytes) { void* p = base + off; off += alignup(bytes); return p; };

  float* deg   = (float*)carve(NN * 4);
  float* dis   = (float*)carve(NN * 4);
  float* snb   = (float*)carve(NN * 4);
  float* s1    = (float*)carve(NN * 4);
  float* norm  = (float*)carve((size_t)ET * 4);
  int*   knn   = (int*)  carve((size_t)NN * KK * 4);

  _Float16* wgh[3], *e1h[3], *e2h[3], *e3h[3];   // all stored transposed [N][K]
  for (int l = 0; l < 3; l++) wgh[l] = (_Float16*)carve(128 * 128 * 2);
  for (int l = 0; l < 3; l++) e1h[l] = (_Float16*)carve(64 * 256 * 2);
  for (int l = 0; l < 3; l++) e2h[l] = (_Float16*)carve(128 * 64 * 2);
  for (int l = 0; l < 3; l++) e3h[l] = (_Float16*)carve(128 * 128 * 2);

  _Float16* acth[2];
  acth[0] = (_Float16*)carve((size_t)NN * HH * 2);
  acth[1] = (_Float16*)carve((size_t)NN * HH * 2);
  float* h_tmp = (float*)carve((size_t)NN * HH * 4);
  float* agg   = (float*)carve((size_t)NN * HH * 4);
  float* n_f32 = (float*)carve((size_t)NN * HH * 4);
  float* e_tmp = (float*)carve((size_t)NN * HH * 4);
  float* t2    = (float*)carve((size_t)NN * HH * 4);

  float* out = (float*)d_out;

  const int B = 256;
  auto gsz = [&](int n) { return (n + B - 1) / B; };

  // ---- weight conversion: f32 [K][N] -> f16 transposed [N][K] ----
  for (int l = 0; l < 3; l++) {
    k_f32_to_h_T<<<gsz(128 * 128), B, 0, stream>>>(wg[l],  wgh[l], 128, 128);
    k_f32_to_h_T<<<gsz(256 * 64),  B, 0, stream>>>(ew1[l], e1h[l], 256, 64);
    k_f32_to_h_T<<<gsz(64 * 128),  B, 0, stream>>>(ew2[l], e2h[l], 64, 128);
    k_f32_to_h_T<<<gsz(128 * 128), B, 0, stream>>>(ew3[l], e3h[l], 128, 128);
  }
  k_f32_to_h<<<gsz(NN * FF), B, 0, stream>>>(x, acth[0], NN * FF);

  // ---- symmetric-normalization coefficients ----
  k_zero<<<gsz(NN), B, 0, stream>>>(deg, NN);
  k_deg<<<gsz(ET), B, 0, stream>>>(dst, ew, deg);
  k_dis<<<gsz(NN), B, 0, stream>>>(deg, dis);
  k_norm<<<gsz(ET), B, 0, stream>>>(src, dst, ew, dis, norm);

  const int strip_blocks = (NN / 16) / 4;             // gemm: wave per 16-row strip
  const int wave4_blocks = (NN / 16 + 3) / 4;         // knn: wave per 16 rows
  const int node_blocks  = NN / 4;                    // edgeconv/rowwise: wave per node/row
  const long long scatT  = (long long)ET * 32;

  for (int l = 0; l < 3; l++) {
    _Float16* a_in  = acth[l & 1];
    _Float16* a_out = acth[(l + 1) & 1];

    // GCN: h = act @ Wg  (WMMA row-strip)
    k_gemm_strip<<<strip_blocks, 128, 0, stream>>>(a_in, wgh[l], h_tmp);
    // aggregate: agg[dst] += norm * h[src]  (+ self loops)
    k_zero<<<gsz(NN * HH), B, 0, stream>>>(agg, NN * HH);
    k_scatter<<<(int)((scatT + B - 1) / B), B, 0, stream>>>(h_tmp, norm, src, dst, agg);
    // n_l = tanh(agg + b), keep f32 + f16 copies
    k_bias_tanh<<<gsz(NN * HH), B, 0, stream>>>(agg, bg[l], n_f32, a_out);

    // dynamic kNN in feature space (WMMA distance tiles + register top-16)
    k_sn<<<node_blocks, 128, 0, stream>>>(n_f32, snb);
    k_knn<<<wave4_blocks, 128, 0, stream>>>(a_out, snb, knn);

    // EdgeConv MLP + max aggregation (WMMA)
    k_edgeconv<<<node_blocks, 128, 0, stream>>>(a_out, knn,
                                                e1h[l], eb1[l],
                                                e2h[l], eb2[l],
                                                e3h[l], eb3[l], e_tmp);

    if (l == 0) {
      k_rownorm<<<node_blocks, 128, 0, stream>>>(e_tmp, n_f32, s1);
    } else if (l == 1) {
      k_add<<<gsz(NN * HH), B, 0, stream>>>(e_tmp, n_f32, t2);
    } else {
      k_final<<<node_blocks, 128, 0, stream>>>(e_tmp, n_f32, t2, s1, out);
    }
  }
}
